// GATConv_50062138802606
// MI455X (gfx1250) — matrix operand order repaired and checked by reading, not compiled, not run on hardware
//
#include <hip/hip_runtime.h>
#include <hip/hip_bf16.h>

// ---------------------------------------------------------------------------
// GAT layer, fused flash-style, bf16 WMMA (v_wmma_f32_16x16x32_bf16)
//   N=4096, M=256, H=8, O=64
//   Round 3: fix async-to-LDS builtin pointer types (AS-qualified v4i*).
// ---------------------------------------------------------------------------

typedef __attribute__((ext_vector_type(16))) __bf16 v16bf;
typedef __attribute__((ext_vector_type(2)))  __bf16 v2bf;
typedef __attribute__((ext_vector_type(8)))  float  v8f;
typedef __attribute__((ext_vector_type(4)))  int    v4i;

#define NN   4096
#define MM   256
#define HH   8
#define OO   64
#define CTOT 1024   // 8*64 projected heads + 512 residual columns

#define AS1 __attribute__((address_space(1)))
#define AS3 __attribute__((address_space(3)))

#if __has_builtin(__builtin_amdgcn_global_load_async_to_lds_b128) && \
    __has_builtin(__builtin_amdgcn_s_wait_asynccnt)
#define HAVE_ASYNC_LDS 1
#else
#define HAVE_ASYNC_LDS 0
#endif

union AFrag { v16bf v; v2bf h[8]; unsigned short u[16]; uint4 q[2]; };
union BFrag { v16bf v; uint4 q[2]; };

__device__ __forceinline__ unsigned short f2bf(float f) {
    union { float f; unsigned int u; } c; c.f = f;
    unsigned int u = c.u;
    return (unsigned short)((u + 0x7FFFu + ((u >> 16) & 1u)) >> 16);
}

// ---------------------------------------------------------------------------
// Kernel 1: x -> bf16, and build transposed combined weight WcT[1024][256]
//   columns 0..511  : w[hd][m][o]  (c = hd*64+o)
//   columns 512..1023: r[m][c-512]
// ---------------------------------------------------------------------------
__global__ void cvt_kernel(const float* __restrict__ x,
                           const float* __restrict__ w,
                           const float* __restrict__ r,
                           unsigned short* __restrict__ xbf,
                           unsigned short* __restrict__ wcT) {
    int idx    = blockIdx.x * blockDim.x + threadIdx.x;
    int stride = gridDim.x * blockDim.x;
    for (int i = idx; i < NN * MM; i += stride)
        xbf[i] = f2bf(x[i]);
    for (int i = idx; i < CTOT * MM; i += stride) {
        int c = i >> 8;          // 0..1023
        int m = i & (MM - 1);    // 0..255
        float v;
        if (c < HH * OO) {
            int hd = c >> 6, o = c & 63;
            v = w[(hd * MM + m) * OO + o];
        } else {
            v = r[m * 512 + (c - 512)];
        }
        wcT[i] = f2bf(v);
    }
}

// ---------------------------------------------------------------------------
// Kernel 2: proj[4096][1024] = x @ [w | r]   (bf16 WMMA, f32 accumulate)
// grid (256, 8), block 256 (8 waves), one 16x16 tile per wave, K=256
// ---------------------------------------------------------------------------
__global__ void proj_gemm(const unsigned short* __restrict__ xbf,
                          const unsigned short* __restrict__ wcT,
                          float* __restrict__ proj) {
    const int lane  = threadIdx.x & 31;
    const int wv    = threadIdx.x >> 5;
    const int ntile = blockIdx.x;              // 0..255
    const int ctile = blockIdx.y * 8 + wv;     // 0..63
    const int rc    = lane & 15;               // A row / B col / D col
    const int half  = lane >> 4;
    const int n0 = ntile * 16;
    const int c0 = ctile * 16;

    v8f acc = {};
    const unsigned short* aRow = xbf + (size_t)(n0 + rc) * MM;
    const unsigned short* bRow = wcT + (size_t)(c0 + rc) * MM;

#pragma unroll
    for (int k = 0; k < MM; k += 32) {
        AFrag a; BFrag b;
        // A 16x32 bf16: lane(half,row) holds K = {k+half*8 .. +7, k+16+half*8 .. +7}
        a.q[0] = *(const uint4*)(aRow + k + half * 8);
        a.q[1] = *(const uint4*)(aRow + k + 16 + half * 8);
        // B 32x16 bf16: lane col=rc holds K = {k+half*8 .. +7, k+16+half*8 .. +7}
        b.q[0] = *(const uint4*)(bRow + k + half * 8);
        b.q[1] = *(const uint4*)(bRow + k + 16 + half * 8);
        acc = __builtin_amdgcn_wmma_f32_16x16x32_bf16(
                  false, a.v, false, b.v, (short)0, acc, false, false);
    }
#pragma unroll
    for (int v = 0; v < 8; ++v)
        proj[(size_t)(n0 + v + 8 * half) * CTOT + c0 + rc] = acc[v];
}

// ---------------------------------------------------------------------------
// Kernel 3: hi[hd][n], hj[hd][n] score dots + transpose h to hbfT[hd][o][n] bf16
// grid (16, 8), block 256; thread -> (hd, n)
// ---------------------------------------------------------------------------
__global__ void scores_kernel(const float* __restrict__ proj,
                              const float* __restrict__ h_i,
                              const float* __restrict__ h_j,
                              float* __restrict__ hi,
                              float* __restrict__ hj,
                              unsigned short* __restrict__ hbfT) {
    int hd = blockIdx.y;
    int n  = blockIdx.x * 256 + threadIdx.x;
    const float* hrow = proj + (size_t)n * CTOT + hd * OO;
    float si = 0.f, sj = 0.f;
#pragma unroll
    for (int o = 0; o < OO; ++o) {
        float v = hrow[o];
        si += v * h_i[hd * OO + o];
        sj += v * h_j[hd * OO + o];
        hbfT[(size_t)(hd * OO + o) * NN + n] = f2bf(v);   // coalesced over n
    }
    hi[hd * NN + n] = si;
    hj[hd * NN + n] = sj;
}

// ---------------------------------------------------------------------------
// Kernel 4: bmax[hd] = max_m hj[hd][m]
// ---------------------------------------------------------------------------
__global__ void bmax_kernel(const float* __restrict__ hj,
                            float* __restrict__ bmax) {
    __shared__ float red[256];
    int hd = blockIdx.x;
    float m = -1e30f;
    for (int n = threadIdx.x; n < NN; n += 256)
        m = fmaxf(m, hj[hd * NN + n]);
    red[threadIdx.x] = m;
    __syncthreads();
    for (int s = 128; s > 0; s >>= 1) {
        if (threadIdx.x < s)
            red[threadIdx.x] = fmaxf(red[threadIdx.x], red[threadIdx.x + s]);
        __syncthreads();
    }
    if (threadIdx.x == 0) bmax[hd] = red[0];
}

// ---------------------------------------------------------------------------
// Kernel 5: fused softmax + aggregation + residual + bias
// All 8 waves of a block share one head; the 64x32-key bf16 value chunk (4 KB)
// is staged once per block into LDS (double-buffered) with async-to-LDS loads,
// prefetching chunk i+1 while chunk i runs exp + 4 WMMAs per wave.
// grid 256, block 256 (8 waves) -> 2048 tiles = 8 heads * 256 n-tiles
// ---------------------------------------------------------------------------
__global__ void agg_kernel(const float* __restrict__ graph,
                           const float* __restrict__ proj,
                           const float* __restrict__ hi,
                           const float* __restrict__ hj,
                           const unsigned short* __restrict__ hbfT,
                           const float* __restrict__ bmax,
                           const float* __restrict__ bias,
                           float* __restrict__ out) {
    __shared__ unsigned short sB[2 * OO * 32];   // 2 x 4 KB double buffer

    const int tid   = threadIdx.x;
    const int lane  = tid & 31;
    const int wv    = tid >> 5;
    const int id    = blockIdx.x * 8 + wv;   // 0..2047
    const int hd    = (blockIdx.x * 8) >> 8; // uniform per block (8 | 256)
    const int ntile = id & 255;              // 0..255
    const int row   = lane & 15;
    const int half  = lane >> 4;
    const int n     = ntile * 16 + row;      // query row owned by this lane (A side)

    // producer mapping: thread t fills 16 B of the key chunk: row o = t/4, seg = t%4
    const int po   = tid >> 2;               // 0..63
    const int pseg = tid & 3;                // 0..3
    const unsigned short* pgsrc =
        hbfT + (size_t)(hd * OO + po) * NN + pseg * 8;
    unsigned short* pdst[2] = { &sB[po * 32 + pseg * 8],
                                &sB[OO * 32 + po * 32 + pseg * 8] };

    const float aval = hi[hd * NN + n];
    // leaky_relu monotone increasing + graph<=0  =>  exact row max bound:
    float t0 = aval + bmax[hd];
    const float smax = fmaxf(t0, 0.2f * t0);

    const float* hjRow = hj + hd * NN;
    const float* gRow  = graph + (size_t)n * NN;

    v8f acc[4] = {};
    float zsum = 0.f;

    // prologue: stage chunk 0 into buffer 0
#if HAVE_ASYNC_LDS
    __builtin_amdgcn_global_load_async_to_lds_b128(
        (AS1 v4i*)(pgsrc), (AS3 v4i*)(pdst[0]), 0, 0);
#else
    *(uint4*)pdst[0] = *(const uint4*)(pgsrc);
#endif

    int buf = 0;
    for (int mb = 0; mb < NN; mb += 32) {
#if HAVE_ASYNC_LDS
        __builtin_amdgcn_s_wait_asynccnt(0);
#endif
        __syncthreads();                     // chunk `buf` visible to all waves

        if (mb + 32 < NN) {                  // prefetch next chunk (uniform branch)
#if HAVE_ASYNC_LDS
            __builtin_amdgcn_global_load_async_to_lds_b128(
                (AS1 v4i*)(pgsrc + mb + 32), (AS3 v4i*)(pdst[buf ^ 1]), 0, 0);
#else
            *(uint4*)pdst[buf ^ 1] = *(const uint4*)(pgsrc + mb + 32);
#endif
        }

        // per-lane key logits: K = {half*8+0..7, 16+half*8+0..7}  (A 16x32 layout)
        float hjv[16], gv[16];
        *(float4*)(hjv + 0)  = *(const float4*)(hjRow + mb + half * 8);
        *(float4*)(hjv + 4)  = *(const float4*)(hjRow + mb + half * 8 + 4);
        *(float4*)(hjv + 8)  = *(const float4*)(hjRow + mb + 16 + half * 8);
        *(float4*)(hjv + 12) = *(const float4*)(hjRow + mb + 16 + half * 8 + 4);
        *(float4*)(gv + 0)   = *(const float4*)(gRow + mb + half * 8);
        *(float4*)(gv + 4)   = *(const float4*)(gRow + mb + half * 8 + 4);
        *(float4*)(gv + 8)   = *(const float4*)(gRow + mb + 16 + half * 8);
        *(float4*)(gv + 12)  = *(const float4*)(gRow + mb + 16 + half * 8 + 4);

        AFrag p;
        float pe[16];
#pragma unroll
        for (int e = 0; e < 16; ++e) {
            float t = aval + hjv[e];
            float s = fmaxf(t, 0.2f * t) + gv[e];   // leaky_relu + graph mask
            pe[e] = __expf(s - smax);
            zsum += pe[e];
        }
#if __has_builtin(__builtin_amdgcn_cvt_pk_bf16_f32)
#pragma unroll
        for (int e = 0; e < 8; ++e)
            p.h[e] = __builtin_amdgcn_cvt_pk_bf16_f32(pe[2 * e], pe[2 * e + 1]);
#else
#pragma unroll
        for (int e = 0; e < 16; ++e)
            p.u[e] = f2bf(pe[e]);
#endif

        const unsigned short* sbuf = &sB[buf * OO * 32];
#pragma unroll
        for (int ot = 0; ot < 4; ++ot) {
            BFrag b;
            const unsigned short* bp = sbuf + (ot * 16 + row) * 32 + half * 8;
            b.q[0] = *(const uint4*)(bp);
            b.q[1] = *(const uint4*)(bp + 16);
            acc[ot] = __builtin_amdgcn_wmma_f32_16x16x32_bf16(
                          false, p.v, false, b.v, (short)0, acc[ot], false, false);
        }

        __syncthreads();                     // all reads of `buf` done before refill
        buf ^= 1;
    }

    // denominator: combine the two K-halves (lane r and lane r+16 share a row)
    float ztot = zsum + __shfl_xor(zsum, 16, 32);

#pragma unroll
    for (int ot = 0; ot < 4; ++ot) {
#pragma unroll
        for (int v = 0; v < 8; ++v) {
            int rowD = v + 8 * half;                 // D layout: VGPR v, half
            float Zr = __shfl(ztot, rowD, 32);       // lane rowD holds Z[rowD]
            int nOut = ntile * 16 + rowD;
            int oc   = ot * 16 + row;                // output column within head
            int c    = hd * OO + oc;
            float y  = acc[ot][v] / Zr
                     + proj[(size_t)nOut * CTOT + 512 + c]   // x@r residual
                     + bias[c];
            out[(size_t)nOut * (HH * OO) + c] = y;
        }
    }
}

// ---------------------------------------------------------------------------
extern "C" void kernel_launch(void* const* d_in, const int* in_sizes, int n_in,
                              void* d_out, int out_size, void* d_ws, size_t ws_size,
                              hipStream_t stream) {
    const float* x     = (const float*)d_in[0];
    const float* graph = (const float*)d_in[1];
    const float* w     = (const float*)d_in[2];
    const float* h_i   = (const float*)d_in[3];
    const float* h_j   = (const float*)d_in[4];
    const float* r     = (const float*)d_in[5];
    const float* bias  = (const float*)d_in[6];
    float* out = (float*)d_out;

    // workspace carve-out (~23.9 MB total)
    char* p = (char*)d_ws;
    unsigned short* xbf  = (unsigned short*)p; p += (size_t)2 * NN * MM;        // 2 MB
    unsigned short* wcT  = (unsigned short*)p; p += (size_t)2 * CTOT * MM;      // 0.5 MB
    float*          proj = (float*)p;          p += (size_t)4 * NN * CTOT;      // 16 MB
    float*          hi   = (float*)p;          p += (size_t)4 * HH * NN;        // 128 KB
    float*          hj   = (float*)p;          p += (size_t)4 * HH * NN;        // 128 KB
    unsigned short* hbfT = (unsigned short*)p; p += (size_t)2 * HH * OO * NN;   // 4 MB
    float*          bmax = (float*)p;          p += 256;

    cvt_kernel<<<256, 256, 0, stream>>>(x, w, r, xbf, wcT);
    proj_gemm<<<dim3(NN / 16, (CTOT / 16) / 8), 256, 0, stream>>>(xbf, wcT, proj);
    scores_kernel<<<dim3(NN / 256, HH), 256, 0, stream>>>(proj, h_i, h_j, hi, hj, hbfT);
    bmax_kernel<<<HH, 256, 0, stream>>>(hj, bmax);
    agg_kernel<<<(HH * (NN / 16)) / 8, 256, 0, stream>>>(graph, proj, hi, hj,
                                                         hbfT, bmax, bias, out);
}
